// GLCMAttentionHead_90048284328251
// MI455X (gfx1250) — compile-verified, matrix operand order checked by reading, not run on hardware
//
#include <hip/hip_runtime.h>
#include <hip/hip_bf16.h>

typedef __attribute__((ext_vector_type(16))) _Float16 v16h;
typedef __attribute__((ext_vector_type(8)))  float    v8f;

#define IMG_H   512
#define IMG_W   512
#define LEVELS  16
#define NBINS   (LEVELS * LEVELS)   // 256
#define ROWS    8                   // rows processed per workgroup

// ---------------------------------------------------------------------------
// Kernel 1: pooled GLCM histogram (4 angles, circular wrap like jnp.roll).
// grid = (numImages, IMG_H/ROWS), block = 256 threads.
// Stages 9 quantized rows (1 halo + 8 work rows) in LDS as u8, scatters into
// a per-WG LDS 256-bin histogram, flushes via one global atomic per bin.
// ---------------------------------------------------------------------------
__global__ void glcm_hist_kernel(const float* __restrict__ x,
                                 unsigned int* __restrict__ counts) {
    __shared__ unsigned char qrow[ROWS + 1][IMG_W];
    __shared__ unsigned int  bins[NBINS];

    const int img  = blockIdx.x;
    const int r0   = blockIdx.y * ROWS;
    const int tid  = threadIdx.x;           // 0..255
    const float* im = x + (size_t)img * IMG_H * IMG_W;

    bins[tid] = 0u;

    // Cooperative load of (ROWS+1) rows, quantized (truncation like the
    // reference's int cast of img*(L-1); inputs are in [0,1)).
    const int n4 = (ROWS + 1) * (IMG_W / 4);          // float4 loads
    for (int li = tid; li < n4; li += 256) {
        const int rr   = li / (IMG_W / 4);            // 0..ROWS
        const int col4 = li % (IMG_W / 4);
        const int row  = (r0 - 1 + rr + IMG_H) % IMG_H;   // circular halo
        const float4 v = ((const float4*)(im + (size_t)row * IMG_W))[col4];
        const int j = col4 * 4;
        qrow[rr][j + 0] = (unsigned char)(v.x * 15.0f);
        qrow[rr][j + 1] = (unsigned char)(v.y * 15.0f);
        qrow[rr][j + 2] = (unsigned char)(v.z * 15.0f);
        qrow[rr][j + 3] = (unsigned char)(v.w * 15.0f);
    }
    __syncthreads();

    // 4096 pixels / 256 threads = 16 pixels each.
    for (int pi = tid; pi < ROWS * IMG_W; pi += 256) {
        const int rr = pi / IMG_W;        // local work row -> LDS row rr+1
        const int j  = pi % IMG_W;
        const int jl = (j + IMG_W - 1) % IMG_W;
        const int jr = (j + 1) % IMG_W;

        const int c   = qrow[rr + 1][j];
        const int lf  = qrow[rr + 1][jl];   // angle 0   : (i, j-1)
        const int upr = qrow[rr][jr];       // angle 45  : (i-1, j+1)
        const int up  = qrow[rr][j];        // angle 90  : (i-1, j)
        const int upl = qrow[rr][jl];       // angle 135 : (i-1, j-1)

        const int base = c * LEVELS;
        atomicAdd(&bins[base + lf],  1u);
        atomicAdd(&bins[base + upr], 1u);
        atomicAdd(&bins[base + up],  1u);
        atomicAdd(&bins[base + upl], 1u);
    }
    __syncthreads();

    atomicAdd(&counts[(size_t)img * NBINS + tid], bins[tid]);
}

// ---------------------------------------------------------------------------
// Kernel 2: entropy of the 256-bin GLCM per image, reduced with one
// v_wmma_f32_16x16x32_f16 (A = 256 terms padded to 16x32 f16, B = ones;
// D row-sums in f32, column N=0 lives on lanes 0 and 16).
// grid = numImages, block = 32 (one full wave: EXEC all ones for WMMA).
// ---------------------------------------------------------------------------
__global__ void glcm_entropy_kernel(const unsigned int* __restrict__ counts,
                                    float* __restrict__ ent) {
    const int img  = blockIdx.x;
    const int lane = threadIdx.x;           // 0..31
    const unsigned int* c = counts + (size_t)img * NBINS;

    // Exact total: every pixel contributes once per angle under jnp.roll.
    const float inv_total = 1.0f / (4.0f * IMG_H * IMG_W);

    v16h a = {};
    #pragma unroll
    for (int k = 0; k < 8; ++k) {
        const float p = (float)c[lane * 8 + k] * inv_total;
        const float term = p * __logf(p + 1e-10f);   // 0 when p == 0
        a[k] = (_Float16)term;                       // slots 8..15 stay zero
    }

    v16h b;
    #pragma unroll
    for (int k = 0; k < 16; ++k) b[k] = (_Float16)1.0f;

    v8f acc = {};
    acc = __builtin_amdgcn_wmma_f32_16x16x32_f16(
        /*neg_a=*/false, a, /*neg_b=*/false, b,
        /*c_mod=*/(short)0, acc, /*reuse_a=*/false, /*reuse_b=*/false);

    // Lane 0 holds D[0..7][0], lane 16 holds D[8..15][0]; with B==1 the row
    // sums cover every A element, i.e. all 256 entropy terms.
    float local = acc[0] + acc[1] + acc[2] + acc[3] +
                  acc[4] + acc[5] + acc[6] + acc[7];
    const float total = __shfl(local, 0, 32) + __shfl(local, 16, 32);
    if (lane == 0) ent[img] = -total;
}

// ---------------------------------------------------------------------------
// Kernel 3: broadcast each image's scalar entropy over its 512x512 slice.
// float4 streaming stores; 65536 float4 per image -> img = gid >> 16.
// ---------------------------------------------------------------------------
__global__ void glcm_bcast_kernel(const float* __restrict__ ent,
                                  float* __restrict__ out) {
    const unsigned int gid = blockIdx.x * 256u + threadIdx.x;  // one float4 each
    const int img = gid >> 16;                                  // /(512*512/4)
    const float e = ent[img];
    ((float4*)out)[gid] = make_float4(e, e, e, e);
}

// ---------------------------------------------------------------------------
extern "C" void kernel_launch(void* const* d_in, const int* in_sizes, int n_in,
                              void* d_out, int out_size, void* d_ws, size_t ws_size,
                              hipStream_t stream) {
    const float* x   = (const float*)d_in[0];
    float*       out = (float*)d_out;

    const int numImages = in_sizes[0] / (IMG_H * IMG_W);   // 8*16 = 128

    unsigned int* counts = (unsigned int*)d_ws;
    float* ent = (float*)((char*)d_ws + (size_t)numImages * NBINS * sizeof(unsigned int));

    hipMemsetAsync(counts, 0, (size_t)numImages * NBINS * sizeof(unsigned int), stream);

    dim3 hgrid(numImages, IMG_H / ROWS);
    glcm_hist_kernel<<<hgrid, 256, 0, stream>>>(x, counts);

    glcm_entropy_kernel<<<numImages, 32, 0, stream>>>(counts, ent);

    const unsigned int nvec4 = (unsigned int)numImages * (IMG_H * IMG_W / 4);
    glcm_bcast_kernel<<<nvec4 / 256, 256, 0, stream>>>(ent, out);
}
